// ConditionGeneration_10282151707544
// MI455X (gfx1250) — matrix operand order repaired and checked by reading, not compiled, not run on hardware
//
#include <hip/hip_runtime.h>
#include <hip/hip_bf16.h>

// ---------------------------------------------------------------------------
// Problem constants (from reference)
// ---------------------------------------------------------------------------
#define B_TOT    4096
#define N_NUM    100
#define N_CAT    100
#define CAT_CNT  100
#define NC       128            // n_cond
#define NK       200            // n_num + n_cat (output inner dim)
#define OUT_ROW  (NC * NK)      // 25600 floats per batch row
#define GN_EPS   1e-5f

typedef _Float16 v16h __attribute__((ext_vector_type(16)));
typedef float    v8f  __attribute__((ext_vector_type(8)));
typedef float    f4   __attribute__((ext_vector_type(4)));

// LDS geometry for the categorical kernel
#define A_STRIDE 132                           // 128 + 4 pad (floats) -> conflict-free frag reads
#define B_STRIDE 136                           // 128 + 8 pad (halves) -> conflict-free transposed writes/reads
#define A_FLOATS (8 * 16 * A_STRIDE)           // 8 waves x 16 rows
#define A_BYTES  (A_FLOATS * 4)                // 67584
#define B_HALVES (NC * B_STRIDE)
#define B_BYTES  (B_HALVES * 2)                // 34816
#define SMEM_BYTES (A_BYTES + 2 * B_BYTES)     // 137216

__device__ __forceinline__ float fast_sigmoid(float x) {
    return 1.0f / (1.0f + __expf(-x));
}

// ---------------------------------------------------------------------------
// Kernel 1: numeric branch. out[b, c, j] = sigmoid(x[b,j]*W[j,c] + bias[j,c])
// Pure store-bandwidth bound; stores are contiguous along j.
// ---------------------------------------------------------------------------
__global__ __launch_bounds__(256) void num_branch_kernel(
        const float* __restrict__ x_num,
        const float* __restrict__ W_num,
        const float* __restrict__ b_num,
        float* __restrict__ out) {
    const int b = blockIdx.x;
    const float* __restrict__ xr = x_num + b * N_NUM;
    float* __restrict__ orow = out + (size_t)b * OUT_ROW;
    for (int e = threadIdx.x; e < NC * N_NUM; e += 256) {
        const int c = e / N_NUM;
        const int j = e - c * N_NUM;
        const float v = xr[j] * W_num[j * NC + c] + b_num[j * NC + c];
        orow[c * NK + j] = fast_sigmoid(v);
    }
}

// ---------------------------------------------------------------------------
// Kernel 2: categorical branch.
// grid = (B/128, N_CAT). Block: 256 threads = 8 waves; wave handles 16 b-rows.
//   Phase 1: stage W_cat[g]^T into LDS as f16 hi/lo split.
//   Phase 2: gather embedding rows, GroupNorm, stage normalized f32 rows.
//   Phase 3: 16x128 x 128x128 GEMM via v_wmma_f32_16x16x32_f16 (hi/lo split,
//            ~f32 accuracy), fused bias + sigmoid, transposed store.
// ---------------------------------------------------------------------------
__global__ __launch_bounds__(256) void cat_branch_kernel(
        const int*   __restrict__ x_cat,
        const float* __restrict__ emb_table,
        const float* __restrict__ gn_w,
        const float* __restrict__ gn_b,
        const float* __restrict__ W_cat,
        const float* __restrict__ b_cat,
        float* __restrict__ out) {
    extern __shared__ char smem_raw[];
    float*    As = (float*)smem_raw;                          // [8][16][A_STRIDE]
    _Float16* Bh = (_Float16*)(smem_raw + A_BYTES);           // [NC][B_STRIDE] = W^T hi
    _Float16* Bl = (_Float16*)(smem_raw + A_BYTES + B_BYTES); // [NC][B_STRIDE] = W^T lo

    const int g    = blockIdx.y;
    const int lane = threadIdx.x & 31;
    const int wv   = threadIdx.x >> 5;

    // ---- Phase 1: stage B = W_cat[g] (K x N) transposed, split f16 hi/lo ----
    for (int idx = threadIdx.x; idx < NC * NC; idx += 256) {
        const int i = idx >> 7;        // K (input feature)
        const int o = idx & 127;       // N (output feature)
        const float w  = W_cat[(size_t)g * NC * NC + idx];
        const _Float16 h = (_Float16)w;
        Bh[(size_t)o * B_STRIDE + i] = h;
        Bl[(size_t)o * B_STRIDE + i] = (_Float16)(w - (float)h);
    }

    // ---- Phase 2: gather + GroupNorm (2 lanes per b-row, 64 cols each) ----
    const int r    = lane >> 1;
    const int half = lane & 1;
    const int brow = blockIdx.x * 128 + wv * 16 + r;
    const int cat  = x_cat[brow * N_CAT + g];
    const float* __restrict__ src = emb_table + ((size_t)g * CAT_CNT + cat) * NC + half * 64;
    float* arow = As + (size_t)(wv * 16 + r) * A_STRIDE + half * 64;

    float s = 0.f, s2 = 0.f;
#pragma unroll
    for (int i = 0; i < 16; ++i) {
        const f4 v = *(const f4*)(src + i * 4);
        *(f4*)(arow + i * 4) = v;
        s  += v.x + v.y + v.z + v.w;
        s2 += v.x * v.x + v.y * v.y + v.z * v.z + v.w * v.w;
    }
    s  += __shfl_xor(s,  1, 32);
    s2 += __shfl_xor(s2, 1, 32);
    const float mean = s * (1.f / NC);
    const float var  = s2 * (1.f / NC) - mean * mean;
    const float rstd = rsqrtf(var + GN_EPS);

    const float* __restrict__ gwp = gn_w + (size_t)g * NC + half * 64;
    const float* __restrict__ gbp = gn_b + (size_t)g * NC + half * 64;
#pragma unroll
    for (int i = 0; i < 16; ++i) {
        const f4 v  = *(const f4*)(arow + i * 4);
        const f4 gw = *(const f4*)(gwp + i * 4);
        const f4 gb = *(const f4*)(gbp + i * 4);
        const f4 nv = (v - mean) * rstd * gw + gb;
        *(f4*)(arow + i * 4) = nv;
    }
    __syncthreads();

    // ---- Phase 3: WMMA GEMM with split-precision f16 ----
    // A frag layout: lane M = lane&15;
    //   lanes 0-15: halves 0-15 -> K+0..15
    //   lanes 16-31: halves 0-15 -> K+16..31
    // B frag: lane column N = lane&15; 16 contiguous K ((lane>>4)*16 offset).
    const float* awave = As + (size_t)wv * 16 * A_STRIDE;
    const int M  = lane & 15;
    const int kb = (lane >> 4) * 16;
    const int kr = (lane >> 4) * 16;

    v8f acc[8];
    const v8f vzero = {};
#pragma unroll
    for (int n = 0; n < 8; ++n) acc[n] = vzero;

#pragma unroll
    for (int kt = 0; kt < 4; ++kt) {
        const int K0 = kt * 32;
        const float* ap = awave + (size_t)M * A_STRIDE + K0 + kb;
        const f4 a0 = *(const f4*)(ap);
        const f4 a1 = *(const f4*)(ap + 4);
        const f4 a2 = *(const f4*)(ap + 8);
        const f4 a3 = *(const f4*)(ap + 12);
        v16h ah, al;
#pragma unroll
        for (int j = 0; j < 4; ++j) {
            float x;
            x = a0[j]; ah[j]      = (_Float16)x; al[j]      = (_Float16)(x - (float)ah[j]);
            x = a1[j]; ah[4 + j]  = (_Float16)x; al[4 + j]  = (_Float16)(x - (float)ah[4 + j]);
            x = a2[j]; ah[8 + j]  = (_Float16)x; al[8 + j]  = (_Float16)(x - (float)ah[8 + j]);
            x = a3[j]; ah[12 + j] = (_Float16)x; al[12 + j] = (_Float16)(x - (float)ah[12 + j]);
        }

#pragma unroll
        for (int n = 0; n < 8; ++n) {
            const int o = n * 16 + (lane & 15);
            const v16h bhf = *(const v16h*)(Bh + (size_t)o * B_STRIDE + K0 + kr);
            const v16h blf = *(const v16h*)(Bl + (size_t)o * B_STRIDE + K0 + kr);
            // D = Ah*Bh + Al*Bh + Ah*Bl + C  (Al*Bl term ~2^-22, dropped)
            acc[n] = __builtin_amdgcn_wmma_f32_16x16x32_f16(
                false, ah, false, bhf, (short)0, acc[n], false, false);
            acc[n] = __builtin_amdgcn_wmma_f32_16x16x32_f16(
                false, al, false, bhf, (short)0, acc[n], false, false);
            acc[n] = __builtin_amdgcn_wmma_f32_16x16x32_f16(
                false, ah, false, blf, (short)0, acc[n], false, false);
        }
    }

    // ---- Epilogue: bias + sigmoid, transposed store out[b, c, 100+g] ----
    // C/D layout: VGPR v -> M = (lane>>4)*8 + v, N = lane&15.
    const int Nc = lane & 15;
    const int Mb = (lane >> 4) * 8;
    const size_t base_b = (size_t)(blockIdx.x * 128 + wv * 16);
#pragma unroll
    for (int n = 0; n < 8; ++n) {
        const float bc = b_cat[(size_t)g * NC + n * 16 + Nc];
        const size_t col = (size_t)(n * 16 + Nc) * NK + N_NUM + g;
#pragma unroll
        for (int v = 0; v < 8; ++v) {
            const float x = acc[n][v] + bc;
            out[(base_b + Mb + v) * OUT_ROW + col] = fast_sigmoid(x);
        }
    }
}

// ---------------------------------------------------------------------------
// Launch
// ---------------------------------------------------------------------------
extern "C" void kernel_launch(void* const* d_in, const int* in_sizes, int n_in,
                              void* d_out, int out_size, void* d_ws, size_t ws_size,
                              hipStream_t stream) {
    const float* x_num   = (const float*)d_in[0];
    const int*   x_cat   = (const int*)  d_in[1];
    const float* W_num   = (const float*)d_in[2];
    const float* b_num   = (const float*)d_in[3];
    const float* emb_tab = (const float*)d_in[4];
    const float* gn_w    = (const float*)d_in[5];
    const float* gn_b    = (const float*)d_in[6];
    const float* W_cat   = (const float*)d_in[7];
    const float* b_cat   = (const float*)d_in[8];
    float* out = (float*)d_out;

    (void)hipFuncSetAttribute((const void*)cat_branch_kernel,
                              hipFuncAttributeMaxDynamicSharedMemorySize,
                              SMEM_BYTES);

    num_branch_kernel<<<B_TOT, 256, 0, stream>>>(x_num, W_num, b_num, out);
    cat_branch_kernel<<<dim3(B_TOT / 128, N_CAT), 256, SMEM_BYTES, stream>>>(
        x_cat, emb_tab, gn_w, gn_b, W_cat, b_cat, out);
}
